// AtomEmbed_30648886624404
// MI455X (gfx1250) — matrix operand order repaired and checked by reading, not compiled, not run on hardware
//
#include <hip/hip_runtime.h>
#include <hip/hip_bf16.h>
#include <stdint.h>

// ---------------------------------------------------------------------------
// AtomEmbed point-transformer for MI455X (gfx1250), wave32 + WMMA f16->f32,
// with Tensor-Data-Mover staging of GEMM A-strips into LDS.
// ---------------------------------------------------------------------------

typedef _Float16 half_t;
typedef __attribute__((ext_vector_type(16))) _Float16 v16h;
typedef __attribute__((ext_vector_type(8)))  float    v8f;
typedef __attribute__((ext_vector_type(4)))  unsigned int uint32x4;
typedef __attribute__((ext_vector_type(8)))  int          int32x8;
typedef __attribute__((ext_vector_type(4)))  int          int32x4;

#define T_BATCH 8
#define NRES    128
#define CATOM   14
#define DIMF    128
#define NPTS    (NRES * CATOM)          // 1792 points per batch element
#define NT      (T_BATCH * NPTS)        // 14336 total point-rows
#define KNN     16
#define INV_SQRT_D 0.0883883476483184f  // 1/sqrt(128)

#if __has_builtin(__builtin_amdgcn_tensor_load_to_lds) && \
    __has_builtin(__builtin_amdgcn_s_wait_tensorcnt)
#define USE_TDM 1
#else
#define USE_TDM 0
#endif

union Frag16 { v16h v; unsigned int u[8]; };

#if USE_TDM
// Issue a TDM 2D tile load: rows x cols (halfs), row stride = cols halfs,
// global tile start -> LDS byte offset. Wave-level op; EXEC ignored.
__device__ __forceinline__ void tdm_load_tile_f16(const half_t* gsrc,
                                                  unsigned int ldsAddr,
                                                  int cols, int rows)
{
    const unsigned long long ga = (unsigned long long)(uintptr_t)gsrc;
    uint32x4 g0;
    g0[0] = 1u;                                       // count=1 (valid), no gather
    g0[1] = ldsAddr;                                  // lds_addr (bytes)
    g0[2] = (unsigned int)(ga & 0xFFFFFFFFull);       // global_addr[31:0]
    g0[3] = (unsigned int)((ga >> 32) & 0x1FFFFFFull) // global_addr[56:32]
          | (2u << 30);                               // type = 2 ("image")
    int32x8 g1;
    g1[0] = (int)(1u << 16);                          // data_size=1 -> 2 bytes
    g1[1] = (int)((unsigned)cols << 16);              // tensor_dim0[15:0]
    g1[2] = (int)((unsigned)rows << 16);              // tensor_dim0 hi=0 | tensor_dim1 lo
    g1[3] = (int)((unsigned)cols << 16);              // tensor_dim1 hi=0 | tile_dim0
    g1[4] = rows;                                     // tile_dim1 | tile_dim2=0
    g1[5] = cols;                                     // tensor_dim0_stride[31:0]
    g1[6] = 0;                                        // stride hi | dim1_stride lo
    g1[7] = 0;
    int32x4 gz = {0, 0, 0, 0};
#if __clang_major__ >= 23
    int32x8 gz8 = {0, 0, 0, 0, 0, 0, 0, 0};
    __builtin_amdgcn_tensor_load_to_lds(g0, g1, gz, gz, gz8, 0);
#else
    __builtin_amdgcn_tensor_load_to_lds(g0, g1, gz, gz, 0);
#endif
}
#endif

// ---------------------------------------------------------------------------
// WMMA GEMM: out = act( A[M,K]_f16 * W[N,K]_f16^T + bias ) (+resid)(*rowScale)
// Block = 256 threads (8 waves) covering a 16x128 C strip: all waves share one
// 16-row A strip, staged into LDS once via the Tensor Data Mover.
// Requires N % 128 == 0, M % 16 == 0, K % 32 == 0, K <= 256.
// ---------------------------------------------------------------------------
__global__ __launch_bounds__(256) void gemm_wmma_kernel(
    const half_t* __restrict__ A, const half_t* __restrict__ W,
    const float* __restrict__ bias, const float* __restrict__ resid,
    const float* __restrict__ rowScale,
    float* __restrict__ outF, half_t* __restrict__ outH,
    int M, int N, int K, int doRelu)
{
    __shared__ half_t Abuf[16 * 256];   // 8 KB A strip (16 rows x K<=256)

    const int nbN = N >> 7;                     // column blocks of 128
    const int tm  = (blockIdx.x / nbN) << 4;    // strip row base
    const int wid = threadIdx.x >> 5;
    const int tn  = ((blockIdx.x % nbN) << 7) + (wid << 4);
    const int lane = threadIdx.x & 31;
    const int lh = lane & 15;
    const int hi = lane >> 4;

#if USE_TDM
    if (wid == 0) {     // wave-uniform: one TDM op per block
        tdm_load_tile_f16(A + (size_t)tm * K,
                          (unsigned int)(uintptr_t)(void*)Abuf, K, 16);
        __builtin_amdgcn_s_wait_tensorcnt(0);
    }
#else
    {   // cooperative fallback: 16*K halfs = 8*K dwords
        const unsigned int* src = (const unsigned int*)(A + (size_t)tm * K);
        unsigned int* dst = (unsigned int*)Abuf;
        const int nd = 8 * K;
        for (int e = threadIdx.x; e < nd; e += 256) dst[e] = src[e];
    }
#endif
    __syncthreads();

    const size_t rowB = (size_t)(tn + lh) * K;  // W row == output column

    v8f acc = {};
    for (int k0 = 0; k0 < K; k0 += 32) {
        if (k0 + 32 < K)
            __builtin_prefetch(W + rowB + k0 + 32, 0, 3);  // global_prefetch_b8
        Frag16 a, b;
#pragma unroll
        for (int j = 0; j < 8; ++j) {
            // A, contiguous k order (k = 16*hi + s) used here, same as B: lanes 0-15 cover K 0..15, lanes 16-31 16..31
            const int ka = k0 + 2 * j + (hi << 4);
            a.u[j] = *(const unsigned int*)(Abuf + lh * K + ka);  // ds_load
            // B, contiguous k order (k = 16*hi + s) used here, same as A: lanes 0-15 cover K 0..15, lanes 16-31 16..31
            const int kb = k0 + 2 * j + (hi << 4);
            b.u[j] = *(const unsigned int*)(W + rowB + kb);
        }
        acc = __builtin_amdgcn_wmma_f32_16x16x32_f16(false, a.v, false, b.v,
                                                     (short)0, acc, false, false);
    }

    const int c = tn + lh;
    const float bv = bias ? bias[c] : 0.0f;
#pragma unroll
    for (int i = 0; i < 8; ++i) {
        const int r = tm + i + (hi << 3);
        float v = acc[i] + bv;
        if (doRelu) v = fmaxf(v, 0.0f);
        if (resid)    v += resid[(size_t)r * N + c];
        if (rowScale) v *= rowScale[r];
        if (outF) outF[(size_t)r * N + c] = v;
        if (outH) outH[(size_t)r * N + c] = (half_t)v;
    }
}

// ---------------------------------------------------------------------------
// f32 -> f16 weight conversion
// ---------------------------------------------------------------------------
__global__ void f32_to_f16_kernel(const float* __restrict__ src,
                                  half_t* __restrict__ dst, int n)
{
    int i = blockIdx.x * blockDim.x + threadIdx.x;
    if (i < n) dst[i] = (half_t)src[i];
}

// pos l1: grid(2ch) -> 256, ReLU, f16   (K=2: VALU)
__global__ void pos_l1_kernel(const float* __restrict__ w, const float* __restrict__ b,
                              half_t* __restrict__ out)
{
    int idx = blockIdx.x * blockDim.x + threadIdx.x;
    if (idx >= NT * 256) return;
    const int r = idx >> 8, o = idx & 255;
    const int i = r % NPTS;
    const float g0 = (float)(i % CATOM) * (1.0f / (CATOM - 1));
    const float g1 = (float)(i / CATOM) * (1.0f / (NRES - 1));
    out[idx] = (half_t)fmaxf(w[o * 2 + 0] * g0 + w[o * 2 + 1] * g1 + b[o], 0.0f);
}

// conv1 l1: xyz(3ch) -> 256, ReLU, f16 (K=3)
__global__ void conv1_l1_kernel(const float* __restrict__ x, const float* __restrict__ w,
                                const float* __restrict__ b, half_t* __restrict__ out)
{
    int idx = blockIdx.x * blockDim.x + threadIdx.x;
    if (idx >= NT * 256) return;
    const int r = idx >> 8, o = idx & 255;
    const float* p = x + (size_t)r * 3;
    out[idx] = (half_t)fmaxf(w[o * 3 + 0] * p[0] + w[o * 3 + 1] * p[1] +
                             w[o * 3 + 2] * p[2] + b[o], 0.0f);
}

// ---------------------------------------------------------------------------
// KNN (computed once; xyz constant across all transformer blocks)
// ---------------------------------------------------------------------------
__global__ void knn_kernel(const float* __restrict__ xyz, const float* __restrict__ mask,
                           int* __restrict__ knn)
{
    const int g = blockIdx.x * blockDim.x + threadIdx.x;
    if (g >= NT) return;
    const int t = g / NPTS;
    const int i = g - t * NPTS;
    const float* P = xyz + (size_t)t * NPTS * 3;
    const float* M = mask + (size_t)t * NPTS;
    const float px = P[i * 3 + 0], py = P[i * 3 + 1], pz = P[i * 3 + 2];
    const float mi = M[i];
    float bd[KNN + 1]; int bi[KNN + 1];
#pragma unroll
    for (int s = 0; s <= KNN; ++s) { bd[s] = 1e30f; bi[s] = 0; }
    for (int j = 0; j < NPTS; ++j) {
        const float dx = px - P[j * 3 + 0];
        const float dy = py - P[j * 3 + 1];
        const float dz = pz - P[j * 3 + 2];
        float d = dx * dx + dy * dy + dz * dz + (1.0f - mi * M[j]) * 1e6f;
        if (d < bd[KNN]) {
            int s = KNN;
            while (s > 0 && bd[s - 1] > d) { bd[s] = bd[s - 1]; bi[s] = bi[s - 1]; --s; }
            bd[s] = d; bi[s] = j;
        }
    }
#pragma unroll
    for (int s = 0; s < KNN; ++s) knn[(size_t)g * KNN + s] = bi[s + 1];
}

// ---------------------------------------------------------------------------
// In-LDS WMMA GEMM stage (32x128x128) for fused attention; 4 waves, 16 tiles.
// mode 0: f16+bias -> outH, 1: relu(f16+bias) -> outH, 2: (f32+bias)/sqrt(d) -> outF
// ---------------------------------------------------------------------------
__device__ __forceinline__ void block_gemm_32x128(
    const half_t* __restrict__ lA, const half_t* __restrict__ W,
    const float* __restrict__ bias, half_t* outH, float* outF, int mode)
{
    const int lane = threadIdx.x & 31;
    const int wid  = threadIdx.x >> 5;
    const int lh = lane & 15, hi = lane >> 4;
    for (int tile = wid; tile < 16; tile += 4) {
        const int tm = (tile >> 3) << 4;
        const int tn = (tile & 7) << 4;
        const int rowA = (tm + lh) * DIMF;
        const size_t rowB = (size_t)(tn + lh) * DIMF;
        v8f acc = {};
        for (int k0 = 0; k0 < DIMF; k0 += 32) {
            Frag16 a, b;
#pragma unroll
            for (int j = 0; j < 8; ++j) {
                const int ka = k0 + 2 * j + (hi << 4);
                a.u[j] = *(const unsigned int*)(lA + rowA + ka);
                const int kb = k0 + 2 * j + (hi << 4);
                b.u[j] = *(const unsigned int*)(W + rowB + kb);
            }
            acc = __builtin_amdgcn_wmma_f32_16x16x32_f16(false, a.v, false, b.v,
                                                         (short)0, acc, false, false);
        }
        const int c = tn + lh;
        const float bv = bias[c];
#pragma unroll
        for (int i = 0; i < 8; ++i) {
            const int r = tm + i + (hi << 3);
            float v = acc[i] + bv;
            if (mode == 2)      outF[r * DIMF + c] = v * INV_SQRT_D;
            else if (mode == 1) outH[r * DIMF + c] = (half_t)fmaxf(v, 0.0f);
            else                outH[r * DIMF + c] = (half_t)v;
        }
    }
}

// ---------------------------------------------------------------------------
// Fused point-transformer attention: 2 points (32 neighbor rows) per block.
// pe = fd2(relu(fd1(rel)));  u = q - k + pe;  a = fg2(relu(fg1(u)))/sqrt(d);
// res = sum_k softmax_k(a) * relu(v + pe)
// ---------------------------------------------------------------------------
__global__ __launch_bounds__(128) void trans_attn_kernel(
    const float* __restrict__ xyz, const int* __restrict__ knn,
    const half_t* __restrict__ q16, const half_t* __restrict__ k16,
    const half_t* __restrict__ v16,
    const float* __restrict__ fd1w, const float* __restrict__ fd1b,
    const half_t* __restrict__ fd2w, const float* __restrict__ fd2b,
    const half_t* __restrict__ fg1w, const float* __restrict__ fg1b,
    const half_t* __restrict__ fg2w, const float* __restrict__ fg2b,
    half_t* __restrict__ res16)
{
    __shared__ half_t Kg[32 * DIMF];   // gathered k rows, later relu(fg1(u))
    __shared__ half_t Vg[32 * DIMF];   // gathered v rows
    __shared__ half_t PE[32 * DIMF];   // positional encoding
    __shared__ half_t AU[32 * DIMF];   // relu(fd1(rel)), then u
    __shared__ float  ATT[32 * DIMF];  // logits -> softmax weights
    __shared__ half_t Qs[2 * DIMF];
    __shared__ float  relS[32 * 4];
    __shared__ int    idxS[32];

    const int tid = threadIdx.x;
    const int g0 = blockIdx.x * 2;
    const int t = g0 / NPTS;
    const int base = t * NPTS;

    if (tid < 32) idxS[tid] = knn[(size_t)g0 * KNN + tid];
    __syncthreads();

    const unsigned int* kq = (const unsigned int*)k16;
    const unsigned int* vq = (const unsigned int*)v16;
    unsigned int* Kgu = (unsigned int*)Kg;
    unsigned int* Vgu = (unsigned int*)Vg;
    for (int e = tid; e < 32 * 64; e += 128) {
        const int r = e >> 6, cu = e & 63;
        const size_t src = (size_t)(base + idxS[r]) * 64 + cu;
        Kgu[e] = kq[src];
        Vgu[e] = vq[src];
    }
    ((unsigned int*)Qs)[tid] =
        ((const unsigned int*)q16)[(size_t)(g0 + (tid >> 6)) * 64 + (tid & 63)];
    if (tid < 32) {
        const int p = tid >> 4;
        const float* pc = xyz + (size_t)(g0 + p) * 3;
        const float* pn = xyz + (size_t)(base + idxS[tid]) * 3;
        relS[tid * 4 + 0] = pc[0] - pn[0];
        relS[tid * 4 + 1] = pc[1] - pn[1];
        relS[tid * 4 + 2] = pc[2] - pn[2];
    }
    __syncthreads();

    for (int e = tid; e < 32 * DIMF; e += 128) {        // relu(fd1(rel))
        const int r = e >> 7, o = e & 127;
        const float v = fd1b[o] + fd1w[o * 3 + 0] * relS[r * 4 + 0]
                                + fd1w[o * 3 + 1] * relS[r * 4 + 1]
                                + fd1w[o * 3 + 2] * relS[r * 4 + 2];
        AU[e] = (half_t)fmaxf(v, 0.0f);
    }
    __syncthreads();

    block_gemm_32x128(AU, fd2w, fd2b, PE, nullptr, 0);  // pe
    __syncthreads();

    for (int e = tid; e < 32 * DIMF; e += 128) {        // u = q - k + pe
        const int r = e >> 7, o = e & 127;
        AU[e] = (half_t)((float)Qs[(r >> 4) * DIMF + o] - (float)Kg[e] + (float)PE[e]);
    }
    __syncthreads();

    block_gemm_32x128(AU, fg1w, fg1b, Kg, nullptr, 1);  // relu(fg1(u))
    __syncthreads();
    block_gemm_32x128(Kg, fg2w, fg2b, nullptr, ATT, 2); // logits / sqrt(d)
    __syncthreads();

    for (int cc = tid; cc < 2 * DIMF; cc += 128) {
        const int p = cc >> 7, o = cc & 127;
        float mx = -1e30f;
#pragma unroll
        for (int j = 0; j < KNN; ++j) mx = fmaxf(mx, ATT[(p * 16 + j) * DIMF + o]);
        float s = 0.0f;
#pragma unroll
        for (int j = 0; j < KNN; ++j) {
            const int rr = (p * 16 + j) * DIMF + o;
            const float ev = __expf(ATT[rr] - mx);
            ATT[rr] = ev; s += ev;
        }
        const float inv = 1.0f / s;
        float acc = 0.0f;
#pragma unroll
        for (int j = 0; j < KNN; ++j) {
            const int rr = (p * 16 + j) * DIMF + o;
            acc += ATT[rr] * inv * fmaxf((float)Vg[rr] + (float)PE[rr], 0.0f);
        }
        res16[(size_t)(g0 + p) * DIMF + o] = (half_t)acc;
    }
}

// ---------------------------------------------------------------------------
// Max pool over the 14 atoms
// ---------------------------------------------------------------------------
__global__ void pool_kernel(const float* __restrict__ h, half_t* __restrict__ pooled)
{
    int idx = blockIdx.x * blockDim.x + threadIdx.x;
    if (idx >= T_BATCH * NRES * DIMF) return;
    const int o = idx & 127;
    const int tn = idx >> 7;
    const int t = tn / NRES, n = tn % NRES;
    const float* bp = h + ((size_t)(t * NPTS + n * CATOM)) * DIMF + o;
    float m = bp[0];
#pragma unroll
    for (int c = 1; c < CATOM; ++c) m = fmaxf(m, bp[c * DIMF]);
    pooled[idx] = (half_t)m;
}

// ---------------------------------------------------------------------------
// Host-side orchestration
// ---------------------------------------------------------------------------
extern "C" void kernel_launch(void* const* d_in, const int* in_sizes, int n_in,
                              void* d_out, int out_size, void* d_ws, size_t ws_size,
                              hipStream_t stream)
{
    (void)in_sizes; (void)n_in; (void)out_size; (void)ws_size;
    // input order (setup_inputs insertion order):
    // 0:x 1:mask | pos:2-5 conv1:6-9 conv2:10-13 conv3:14-17 conv4:18-21
    // trans1:22-36 trans2:37-51 trans3:52-66 | lin1:67-68 lin2:69-70
    const float* x    = (const float*)d_in[0];    // xyz: [T, NPTS, 3]
    const float* mask = (const float*)d_in[1];    // [T, NPTS]
    auto F = [&](int i) { return (const float*)d_in[i]; };

    char* ws = (char*)d_ws;
    size_t off = 0;
    auto alloc = [&](size_t bytes) -> char* {
        char* p = ws + off;
        off = (off + bytes + 255) & ~(size_t)255;
        return p;
    };

    int cvtIdx[40], cvtN[40], nc = 0;
    auto addc = [&](int i, int n) { cvtIdx[nc] = i; cvtN[nc] = n; ++nc; };
    addc(4, 128 * 256);                       // 0: pos l2
    addc(8, 128 * 256);                       // 1: conv1 l2
    addc(10, 256 * 128); addc(12, 128 * 256); // 2,3: conv2
    addc(14, 256 * 128); addc(16, 128 * 256); // 4,5: conv3
    addc(18, 256 * 128); addc(20, 128 * 256); // 6,7: conv4
    for (int b = 0; b < 3; ++b) {             // 8+b*8: fc1 fc2 fd2 fg1 fg2 wq wk wv
        const int t0 = 22 + b * 15;
        addc(t0 + 0, 128 * 128); addc(t0 + 2, 128 * 128); addc(t0 + 6, 128 * 128);
        addc(t0 + 8, 128 * 128); addc(t0 + 10, 128 * 128);
        addc(t0 + 12, 128 * 128); addc(t0 + 13, 128 * 128); addc(t0 + 14, 128 * 128);
    }
    addc(67, 256 * 128);                      // 32: lin1
    addc(69, 128 * 256);                      // 33: lin2
    half_t* wh[40];
    for (int i = 0; i < nc; ++i) {
        wh[i] = (half_t*)alloc((size_t)cvtN[i] * sizeof(half_t));
        f32_to_f16_kernel<<<(cvtN[i] + 255) / 256, 256, 0, stream>>>(F(cvtIdx[i]), wh[i], cvtN[i]);
    }

    float*  hF_a = (float*)alloc((size_t)NT * DIMF * 4);
    float*  hF_b = (float*)alloc((size_t)NT * DIMF * 4);
    half_t* hH_a = (half_t*)alloc((size_t)NT * DIMF * 2);
    half_t* hH_b = (half_t*)alloc((size_t)NT * DIMF * 2);
    half_t* t256 = (half_t*)alloc((size_t)NT * 256 * 2);
    float*  posF = (float*)alloc((size_t)NT * DIMF * 4);
    half_t* x116 = (half_t*)alloc((size_t)NT * DIMF * 2);
    half_t* q16  = (half_t*)alloc((size_t)NT * DIMF * 2);
    half_t* k16  = (half_t*)alloc((size_t)NT * DIMF * 2);
    half_t* v16  = (half_t*)alloc((size_t)NT * DIMF * 2);
    half_t* r16  = (half_t*)alloc((size_t)NT * DIMF * 2);
    int*    knn  = (int*)alloc((size_t)NT * KNN * 4);
    half_t* pool16 = (half_t*)alloc((size_t)T_BATCH * NRES * DIMF * 2);
    half_t* l1o16  = (half_t*)alloc((size_t)T_BATCH * NRES * 256 * 2);

    auto gemm = [&](const half_t* A, const half_t* Wt, const float* bias,
                    const float* resid, const float* rowScale,
                    float* outF, half_t* outH, int M, int N, int K, int relu) {
        const int blocks = (M / 16) * (N / 128);
        gemm_wmma_kernel<<<blocks, 256, 0, stream>>>(
            A, Wt, bias, resid, rowScale, outF, outH, M, N, K, relu);
    };

    knn_kernel<<<(NT + 127) / 128, 128, 0, stream>>>(x, mask, knn);

    // pos = conv(grid): l1 VALU, l2 WMMA
    pos_l1_kernel<<<(NT * 256 + 255) / 256, 256, 0, stream>>>(F(2), F(3), t256);
    gemm(t256, wh[0], F(5), nullptr, nullptr, posF, nullptr, NT, 128, 256, 1);

    // h0 = conv1(xyz) + pos
    conv1_l1_kernel<<<(NT * 256 + 255) / 256, 256, 0, stream>>>(x, F(6), F(7), t256);
    gemm(t256, wh[1], F(9), posF, nullptr, hF_a, hH_a, NT, 128, 256, 1);

    float*  curF = hF_a; half_t* curH = hH_a;
    float*  altF = hF_b; half_t* altH = hH_b;

    const int convL1w[3] = {2, 4, 6}, convL2w[3] = {3, 5, 7};
    const int convL1b[3] = {11, 15, 19}, convL2b[3] = {13, 17, 21};

    for (int b = 0; b < 3; ++b) {
        const int t0 = 22 + b * 15;
        const int wb = 8 + b * 8;
        // x1 = fc1(h); q/k/v = x1 @ W{q,k,v}^T
        gemm(curH, wh[wb + 0], F(t0 + 1), nullptr, nullptr, nullptr, x116, NT, 128, 128, 0);
        gemm(x116, wh[wb + 5], nullptr, nullptr, nullptr, nullptr, q16, NT, 128, 128, 0);
        gemm(x116, wh[wb + 6], nullptr, nullptr, nullptr, nullptr, k16, NT, 128, 128, 0);
        gemm(x116, wh[wb + 7], nullptr, nullptr, nullptr, nullptr, v16, NT, 128, 128, 0);
        // fused attention
        trans_attn_kernel<<<NT / 2, 128, 0, stream>>>(
            x, knn, q16, k16, v16,
            F(t0 + 4), F(t0 + 5), wh[wb + 2], F(t0 + 7),
            wh[wb + 3], F(t0 + 9), wh[wb + 4], F(t0 + 11), r16);
        // h = (fc2(res) + h) * mask
        gemm(r16, wh[wb + 1], F(t0 + 3), curF, mask, altF, altH, NT, 128, 128, 0);
        { float* tf = curF; curF = altF; altF = tf; }
        { half_t* th = curH; curH = altH; altH = th; }
        // h = conv_{b+2}(h) + h   (mask applied on the last conv: h*mp)
        gemm(curH, wh[convL1w[b]], F(convL1b[b]), nullptr, nullptr,
             nullptr, t256, NT, 256, 128, 1);
        gemm(t256, wh[convL2w[b]], F(convL2b[b]), curF, (b == 2) ? mask : nullptr,
             altF, altH, NT, 128, 256, 1);
        { float* tf = curF; curF = altF; altF = tf; }
        { half_t* th = curH; curH = altH; altH = th; }
    }

    pool_kernel<<<(T_BATCH * NRES * DIMF + 255) / 256, 256, 0, stream>>>(curF, pool16);
    gemm(pool16, wh[32], F(68), nullptr, nullptr, nullptr, l1o16,
         T_BATCH * NRES, 256, 128, 1);
    gemm(l1o16, wh[33], F(70), nullptr, nullptr, (float*)d_out, nullptr,
         T_BATCH * NRES, 128, 256, 0);
}
